// BaseLayerGate_23278722744540
// MI455X (gfx1250) — matrix operand order repaired and checked by reading, not compiled, not run on hardware
//
#include <hip/hip_runtime.h>
#include <hip/hip_bf16.h>
#include <stdint.h>

typedef __attribute__((ext_vector_type(16))) __bf16    v16bf;
typedef __attribute__((ext_vector_type(8)))  float     v8f;
typedef __attribute__((ext_vector_type(4)))  float     f4;
typedef __attribute__((ext_vector_type(4)))  unsigned  u4;

#define N_TOKENS  65536
#define D_MODEL   1024
#define N_EXPERTS 64
#define CAP       1024      // N_TOKENS / N_EXPERTS
#define NBINS     4096
#define NBLK      256       // 256 blocks * 256 threads = 65536 tokens

// ---------- helpers ----------
__device__ __forceinline__ uint16_t f2bf(float f) {
  unsigned u = __float_as_uint(f);
  unsigned r = u + 0x7FFFu + ((u >> 16) & 1u);   // round-to-nearest-even
  return (uint16_t)(r >> 16);
}
__device__ __forceinline__ unsigned pk(float a, float b) {
  return (unsigned)f2bf(a) | ((unsigned)f2bf(b) << 16);
}
// monotonic float -> uint key (bigger key == bigger float)
__device__ __forceinline__ unsigned fkey(float f) {
  unsigned u = __float_as_uint(f);
  return (u & 0x80000000u) ? ~u : (u | 0x80000000u);
}

// ---------- phase 0: convert gate weights to bf16 (128 KB, stays L2-resident) ----------
__global__ __launch_bounds__(256)
void wcvt_kernel(const float* __restrict__ w, uint16_t* __restrict__ wb) {
  int i = blockIdx.x * 256 + threadIdx.x;       // 65536 total
  wb[i] = f2bf(w[i]);
}

// ---------- phase 1: gate GEMM via bf16 WMMA, store scores transposed [E, N] ----------
// Block = 256 thr = 8 waves; wave owns a 16-token x 64-expert strip (4 WMMA tiles).
// A (features, 256 MB, read-once) -> non-temporal loads, prefetched 64-K ahead.
// B (bf16 weights, 128 KB, L2-resident) -> ping-pong double buffer, loaded a
// half-iteration ahead so no WMMA waits on an in-flight L2 round trip.
struct Araw { f4 r[4]; };
struct Apk  { union { unsigned u[8]; v16bf v; } m; };
struct Btile { union { u4 q[2]; v16bf v; } t[4]; };

__device__ __forceinline__ void loadA_nt(const float* p, Araw& a) {
  a.r[0] = __builtin_nontemporal_load((const f4*)(p));
  a.r[1] = __builtin_nontemporal_load((const f4*)(p + 4));
  a.r[2] = __builtin_nontemporal_load((const f4*)(p + 16));
  a.r[3] = __builtin_nontemporal_load((const f4*)(p + 20));
}
__device__ __forceinline__ void packA(const Araw& a, Apk& o) {
#pragma unroll
  for (int j = 0; j < 4; ++j) {
    o.m.u[2 * j]     = pk(a.r[j][0], a.r[j][1]);
    o.m.u[2 * j + 1] = pk(a.r[j][2], a.r[j][3]);
  }
}
__device__ __forceinline__ void loadB(const uint16_t* bp, Btile& b) {
#pragma unroll
  for (int t = 0; t < 4; ++t) {
    b.t[t].q[0] = *(const u4*)(bp + t * 16 * D_MODEL);
    b.t[t].q[1] = *(const u4*)(bp + t * 16 * D_MODEL + 16);
  }
}

__global__ __launch_bounds__(256)
void gate_gemm_kernel(const float* __restrict__ feat,
                      const uint16_t* __restrict__ wb,
                      float* __restrict__ scoresT) {
  const int wave  = threadIdx.x >> 5;           // 0..7
  const int lane  = threadIdx.x & 31;
  const int half  = lane >> 4;                  // 0/1
  const int l15   = lane & 15;
  const int mBase = blockIdx.x * 128 + wave * 16;

  // A layout (16-bit A 16x32): lane<16 -> K {0..7,16..23}, lane>=16 -> K {8..15,24..31}
  const float* aPtr = feat + (size_t)(mBase + l15) * D_MODEL + half * 8;
  // B layout (32x16): lane&15 = expert col, lane<16 -> K {0..7,16..23}, lane>=16 -> K {8..15,24..31}
  const uint16_t* bBase = wb + (size_t)l15 * D_MODEL + half * 8;

  v8f acc0 = {}, acc1 = {}, acc2 = {}, acc3 = {};
  Araw ra, rb;
  Btile Ba, Bb;
  Apk Aa, Ab;

  // prologue
  loadA_nt(aPtr, ra);          // A raw for K 0..31
  loadA_nt(aPtr + 32, rb);     // A raw for K 32..63
  loadB(bBase, Ba);            // B for K 0..31

  // steady state: k = 0, 64, ..., 896  (last 64-K chunk peeled below)
  for (int k = 0; k < D_MODEL - 64; k += 64) {
    packA(ra, Aa);
    loadB(bBase + k + 32, Bb);         // B for second half, in flight under WMMAs
    loadA_nt(aPtr + k + 64, ra);       // A raw for next iter, first half
    acc0 = __builtin_amdgcn_wmma_f32_16x16x32_bf16(false, Aa.m.v, false, Ba.t[0].v, (short)0, acc0, false, false);
    acc1 = __builtin_amdgcn_wmma_f32_16x16x32_bf16(false, Aa.m.v, false, Ba.t[1].v, (short)0, acc1, false, false);
    acc2 = __builtin_amdgcn_wmma_f32_16x16x32_bf16(false, Aa.m.v, false, Ba.t[2].v, (short)0, acc2, false, false);
    acc3 = __builtin_amdgcn_wmma_f32_16x16x32_bf16(false, Aa.m.v, false, Ba.t[3].v, (short)0, acc3, false, false);

    packA(rb, Ab);
    loadB(bBase + k + 64, Ba);         // B for next iter, first half
    loadA_nt(aPtr + k + 96, rb);       // A raw for next iter, second half
    acc0 = __builtin_amdgcn_wmma_f32_16x16x32_bf16(false, Ab.m.v, false, Bb.t[0].v, (short)0, acc0, false, false);
    acc1 = __builtin_amdgcn_wmma_f32_16x16x32_bf16(false, Ab.m.v, false, Bb.t[1].v, (short)0, acc1, false, false);
    acc2 = __builtin_amdgcn_wmma_f32_16x16x32_bf16(false, Ab.m.v, false, Bb.t[2].v, (short)0, acc2, false, false);
    ac3:
    acc3 = __builtin_amdgcn_wmma_f32_16x16x32_bf16(false, Ab.m.v, false, Bb.t[3].v, (short)0, acc3, false, false);
  }

  // epilogue: k = 960 (K 960..1023)
  {
    packA(ra, Aa);
    loadB(bBase + (D_MODEL - 32), Bb);
    acc0 = __builtin_amdgcn_wmma_f32_16x16x32_bf16(false, Aa.m.v, false, Ba.t[0].v, (short)0, acc0, false, false);
    acc1 = __builtin_amdgcn_wmma_f32_16x16x32_bf16(false, Aa.m.v, false, Ba.t[1].v, (short)0, acc1, false, false);
    acc2 = __builtin_amdgcn_wmma_f32_16x16x32_bf16(false, Aa.m.v, false, Ba.t[2].v, (short)0, acc2, false, false);
    acc3 = __builtin_amdgcn_wmma_f32_16x16x32_bf16(false, Aa.m.v, false, Ba.t[3].v, (short)0, acc3, false, false);
    packA(rb, Ab);
    acc0 = __builtin_amdgcn_wmma_f32_16x16x32_bf16(false, Ab.m.v, false, Bb.t[0].v, (short)0, acc0, false, false);
    acc1 = __builtin_amdgcn_wmma_f32_16x16x32_bf16(false, Ab.m.v, false, Bb.t[1].v, (short)0, acc1, false, false);
    acc2 = __builtin_amdgcn_wmma_f32_16x16x32_bf16(false, Ab.m.v, false, Bb.t[2].v, (short)0, acc2, false, false);
    acc3 = __builtin_amdgcn_wmma_f32_16x16x32_bf16(false, Ab.m.v, false, Bb.t[3].v, (short)0, acc3, false, false);
  }

  // D layout: VGPR r, lane l -> token mBase + (l>>4)*8 + r, expert tile col l&15.
  const int tokBase = mBase + half * 8;
  v8f accs[4] = {acc0, acc1, acc2, acc3};
#pragma unroll
  for (int t = 0; t < 4; ++t) {
    int expert = t * 16 + l15;
    float* o = scoresT + (size_t)expert * N_TOKENS + tokBase;
    f4 lo = {accs[t][0], accs[t][1], accs[t][2], accs[t][3]};
    f4 hi = {accs[t][4], accs[t][5], accs[t][6], accs[t][7]};
    *(f4*)o       = lo;
    *(f4*)(o + 4) = hi;
  }
}

// ---------- phase 2: per-expert radix select (columns are contiguous, L2-resident) ----------
__global__ __launch_bounds__(256)
void hist_kernel(const float* __restrict__ col, const unsigned* __restrict__ assigned,
                 unsigned* __restrict__ hist) {
  __shared__ unsigned h[NBINS];
  for (int i = threadIdx.x; i < NBINS; i += 256) h[i] = 0;
  __syncthreads();
  int tok = blockIdx.x * 256 + threadIdx.x;
  if (!assigned[tok]) atomicAdd(&h[fkey(col[tok]) >> 20], 1u);
  __syncthreads();
  for (int i = threadIdx.x; i < NBINS; i += 256)
    if (h[i]) atomicAdd(&hist[i], h[i]);
}

// single block of 256: find threshold bin T (descending), need-from-bin, count-above
__global__ __launch_bounds__(256)
void threshold_kernel(const unsigned* __restrict__ hist, int* __restrict__ scal) {
  __shared__ unsigned gs[256];
  int t = threadIdx.x;
  unsigned s = 0;
  for (int i = 0; i < 16; ++i) s += hist[t * 16 + i];
  gs[t] = s;
  __syncthreads();
  if (t == 0) {
    unsigned cum = 0;
    int g = 255;
    for (; g > 0; --g) { if (cum + gs[g] >= CAP) break; cum += gs[g]; }
    int bin = g * 16 + 15;
    for (; bin > g * 16; --bin) { unsigned c = hist[bin]; if (cum + c >= CAP) break; cum += c; }
    scal[0] = bin;              // threshold bin
    scal[1] = CAP - (int)cum;   // tokens still needed from threshold bin
    scal[2] = (int)cum;         // tokens strictly above threshold bin
  }
}

__global__ __launch_bounds__(256)
void count_kernel(const float* __restrict__ col, const unsigned* __restrict__ assigned,
                  const int* __restrict__ scal,
                  unsigned* __restrict__ blkA, unsigned* __restrict__ blkB) {
  __shared__ unsigned sa[256], sb[256];
  int tok = blockIdx.x * 256 + threadIdx.x;
  unsigned T = (unsigned)scal[0];
  unsigned above = 0, inb = 0;
  if (!assigned[tok]) {
    unsigned bin = fkey(col[tok]) >> 20;
    above = (bin > T); inb = (bin == T);
  }
  sa[threadIdx.x] = above; sb[threadIdx.x] = inb;
  __syncthreads();
  for (int off = 128; off > 0; off >>= 1) {
    if (threadIdx.x < off) {
      sa[threadIdx.x] += sa[threadIdx.x + off];
      sb[threadIdx.x] += sb[threadIdx.x + off];
    }
    __syncthreads();
  }
  if (threadIdx.x == 0) { blkA[blockIdx.x] = sa[0]; blkB[blockIdx.x] = sb[0]; }
}

// single block: exclusive scan of 256 per-block counts (O(256^2), trivial)
__global__ __launch_bounds__(256)
void scan_kernel(const unsigned* __restrict__ blkA, const unsigned* __restrict__ blkB,
                 unsigned* __restrict__ offA, unsigned* __restrict__ offB) {
  __shared__ unsigned a[256], b[256];
  int t = threadIdx.x;
  a[t] = blkA[t]; b[t] = blkB[t];
  __syncthreads();
  unsigned sA = 0, sB = 0;
  for (int j = 0; j < t; ++j) { sA += a[j]; sB += b[j]; }
  offA[t] = sA; offB[t] = sB;
}

__global__ __launch_bounds__(256)
void assign_kernel(const float* __restrict__ col, unsigned* __restrict__ assigned,
                   const int* __restrict__ scal,
                   const unsigned* __restrict__ offA, const unsigned* __restrict__ offB,
                   float* __restrict__ outSort, int expert) {
  __shared__ unsigned pa[256], pb[256];
  int tid = threadIdx.x;
  int tok = blockIdx.x * 256 + tid;
  unsigned T = (unsigned)scal[0];
  int need = scal[1], totalAbove = scal[2];
  unsigned above = 0, inb = 0;
  if (!assigned[tok]) {
    unsigned bin = fkey(col[tok]) >> 20;
    above = (bin > T); inb = (bin == T);
  }
  pa[tid] = above; pb[tid] = inb;
  __syncthreads();
  for (int off = 1; off < 256; off <<= 1) {          // Hillis-Steele inclusive scan
    unsigned xa = (tid >= off) ? pa[tid - off] : 0;
    unsigned xb = (tid >= off) ? pb[tid - off] : 0;
    __syncthreads();
    pa[tid] += xa; pb[tid] += xb;
    __syncthreads();
  }
  unsigned exA = pa[tid] - above;
  unsigned exB = pb[tid] - inb;
  if (above) {
    unsigned pos = offA[blockIdx.x] + exA;
    assigned[tok] = 1;
    outSort[expert * CAP + pos] = (float)tok;
  } else if (inb) {
    unsigned r = offB[blockIdx.x] + exB;
    if ((int)r < need) {
      assigned[tok] = 1;
      outSort[expert * CAP + totalAbove + r] = (float)tok;
    }
  }
}

// ---------- phase 3: constant outputs (splits = 1024, straight-through probs = 1.0) ----------
__global__ __launch_bounds__(256)
void init_out_kernel(float* __restrict__ out) {
  int i = blockIdx.x * 256 + threadIdx.x;
  if (i < 2 * N_EXPERTS)
    __builtin_nontemporal_store((float)CAP, &out[N_TOKENS + i]);          // splits
  if (i < N_TOKENS)
    __builtin_nontemporal_store(1.0f, &out[N_TOKENS + 2 * N_EXPERTS + i]); // probs
}

extern "C" void kernel_launch(void* const* d_in, const int* in_sizes, int n_in,
                              void* d_out, int out_size, void* d_ws, size_t ws_size,
                              hipStream_t stream) {
  const float* feat = (const float*)d_in[0];   // [65536, 1024] f32
  const float* wg   = (const float*)d_in[1];   // [64, 1024]    f32
  float* out = (float*)d_out;                  // 65536 + 64 + 64 + 65536 floats
  char* ws = (char*)d_ws;

  // workspace layout
  float*    scoresT  = (float*)ws;                               // 16 MB  [E, N]
  uint16_t* wb       = (uint16_t*)(ws + 16777216);               // 128 KB bf16 weights
  unsigned* assigned = (unsigned*)(ws + 16908288);               // 256 KB
  unsigned* hist     = (unsigned*)(ws + 17170432);               // 16 KB
  unsigned* blkA     = (unsigned*)(ws + 17186816);               // 1 KB
  unsigned* blkB     = (unsigned*)(ws + 17187840);               // 1 KB
  unsigned* offA     = (unsigned*)(ws + 17188864);               // 1 KB
  unsigned* offB     = (unsigned*)(ws + 17189888);               // 1 KB
  int*      scal     = (int*)(ws + 17190912);                    // 16 B

  hipMemsetAsync(assigned, 0, N_TOKENS * sizeof(unsigned), stream);
  wcvt_kernel<<<256, 256, 0, stream>>>(wg, wb);
  gate_gemm_kernel<<<N_TOKENS / 128, 256, 0, stream>>>(feat, wb, scoresT);
  init_out_kernel<<<NBLK, 256, 0, stream>>>(out);

  for (int e = 0; e < N_EXPERTS; ++e) {
    const float* col = scoresT + (size_t)e * N_TOKENS;
    hipMemsetAsync(hist, 0, NBINS * sizeof(unsigned), stream);
    hist_kernel<<<NBLK, 256, 0, stream>>>(col, assigned, hist);
    threshold_kernel<<<1, 256, 0, stream>>>(hist, scal);
    count_kernel<<<NBLK, 256, 0, stream>>>(col, assigned, scal, blkA, blkB);
    scan_kernel<<<1, 256, 0, stream>>>(blkA, blkB, offA, offB);
    assign_kernel<<<NBLK, 256, 0, stream>>>(col, assigned, scal, offA, offB, out, e);
  }
}
